// SelfAttentionEncoder_87265145520556
// MI455X (gfx1250) — matrix operand order repaired and checked by reading, not compiled, not run on hardware
//
#include <hip/hip_runtime.h>
#include <hip/hip_bf16.h>
#include <stdint.h>

#define S_   1024
#define D_   1024
#define B_   4
#define H_   16
#define HD_  64
#define NB_  11
#define CLIP_ 5

typedef __bf16 bf16;
typedef __attribute__((ext_vector_type(8)))  __bf16 bf16x8;
typedef __attribute__((ext_vector_type(16))) __bf16 bf16x16;
typedef __attribute__((ext_vector_type(8)))  float  floatx8;
typedef __attribute__((ext_vector_type(4)))  uint32_t uint32x4;
typedef __attribute__((ext_vector_type(8)))  uint32_t uint32x8;

static __device__ __forceinline__ floatx8 wmma_bf16(bf16x16 a, bf16x16 b, floatx8 c) {
  // D(f32 16x16) = A(bf16 16x32) * B(bf16 32x16) + C
  return __builtin_amdgcn_wmma_f32_16x16x32_bf16(false, a, false, b, (short)0, c, false, false);
}
static __device__ __forceinline__ bf16x16 cat8(bf16x8 lo, bf16x8 hi) {
  return __builtin_shufflevector(lo, hi, 0,1,2,3,4,5,6,7,8,9,10,11,12,13,14,15);
}
static __device__ __forceinline__ float rmax16(float x) {
#pragma unroll
  for (int o = 8; o > 0; o >>= 1) x = fmaxf(x, __shfl_xor(x, o, 16));
  return x;
}
static __device__ __forceinline__ float rsum16(float x) {
#pragma unroll
  for (int o = 8; o > 0; o >>= 1) x += __shfl_xor(x, o, 16);
  return x;
}

// --- Tensor Data Mover: 2D tile (bf16) global -> LDS with row padding -------
// tile_x elements per row, rows rows, global row stride = stride_elems.
// Pad: after every 16 DWORDs (64B row) insert 8 DWORDs (32B) -> 96B LDS row.
static __device__ __forceinline__ void tdm_load_2d_bf16(uint32_t lds_addr,
                                                        const void* gaddr,
                                                        uint32_t tile_x,
                                                        uint32_t rows,
                                                        uint32_t stride_elems) {
  const uint64_t ga = (uint64_t)(uintptr_t)gaddr;
  uint32x4 g0;
  g0[0] = 1u;                                       // count=1, user descriptor
  g0[1] = lds_addr;                                 // LDS byte address
  g0[2] = (uint32_t)(ga & 0xFFFFFFFFu);             // global_addr[31:0]
  g0[3] = (uint32_t)((ga >> 32) & 0x01FFFFFFu)      // global_addr[56:32]
        | (2u << 30);                               // type = 2 ("image")
  uint32x8 g1;
  g1[0] = (1u << 16)                                // data_size = 2 bytes
        | (1u << 20)                                // pad_enable
        | (3u << 22)                                // pad_interval: 16 DWORDs
        | (7u << 25);                               // pad_amount: 8 DWORDs
  const uint32_t td0 = 0x40000000u, td1 = 0x40000000u;  // huge dims: no OOB clamp
  g1[1] = (td0 & 0xFFFFu) << 16;                    // tensor_dim0 lo16
  g1[2] = (td0 >> 16) | ((td1 & 0xFFFFu) << 16);    // tensor_dim0 hi / dim1 lo
  g1[3] = (td1 >> 16) | (tile_x << 16);             // tensor_dim1 hi / tile_dim0
  g1[4] = rows;                                     // tile_dim1 (tile_dim2 = 0)
  g1[5] = stride_elems;                             // tensor_dim0_stride lo32
  g1[6] = 0u;
  g1[7] = 0u;
  asm volatile("tensor_load_to_lds %0, %1" :: "s"(g0), "s"(g1) : "memory");
}

// ---------------- elementwise f32 -> bf16 ----------------
__global__ void cvt_f32_bf16_kernel(const float* __restrict__ in, bf16* __restrict__ out, int n) {
  int i = blockIdx.x * blockDim.x + threadIdx.x;
  if (i < n) out[i] = (bf16)in[i];
}

// ---------------- W[K][N] f32 -> Wt[N][K] bf16 (tiled transpose) ----------------
__global__ __launch_bounds__(256) void wt_convert_kernel(const float* __restrict__ W,
                                                         bf16* __restrict__ Wt,
                                                         int Kd, int Nd) {
  __shared__ float t[32][33];
  const int n0 = blockIdx.x * 32, k0 = blockIdx.y * 32;
  const int tx = threadIdx.x, ty = threadIdx.y;
#pragma unroll
  for (int rr = 0; rr < 4; ++rr)
    t[ty + rr * 8][tx] = W[(size_t)(k0 + ty + rr * 8) * Nd + n0 + tx];
  __syncthreads();
#pragma unroll
  for (int rr = 0; rr < 4; ++rr)
    Wt[(size_t)(n0 + ty + rr * 8) * Kd + k0 + tx] = (bf16)t[tx][ty + rr * 8];
}

// ---------------- V[B,S,D] bf16 -> Vt[B,D,S] bf16 ----------------
__global__ __launch_bounds__(256) void vt_kernel(const bf16* __restrict__ V, bf16* __restrict__ Vt) {
  __shared__ bf16 t[32][34];
  const int b = blockIdx.z;
  const int i0 = blockIdx.x * 32, d0 = blockIdx.y * 32;
  const int tx = threadIdx.x, ty = threadIdx.y;
#pragma unroll
  for (int rr = 0; rr < 4; ++rr)
    t[ty + rr * 8][tx] = V[((size_t)b * S_ + i0 + ty + rr * 8) * D_ + d0 + tx];
  __syncthreads();
#pragma unroll
  for (int rr = 0; rr < 4; ++rr)
    Vt[((size_t)b * D_ + d0 + ty + rr * 8) * S_ + i0 + tx] = t[tx][ty + rr * 8];
}

// ---------------- Qbias[b,h,i,r] = Q[b,i,h,:] . key_bias[r,:] ----------------
__global__ __launch_bounds__(256) void qbias_kernel(const bf16* __restrict__ Qh,
                                                    const float* __restrict__ key_bias,
                                                    float* __restrict__ Qbias) {
  const int t = blockIdx.x * 256 + threadIdx.x;
  if (t >= B_ * S_ * H_) return;
  const int h = t % H_;
  const int i = (t / H_) % S_;
  const int b = t / (H_ * S_);
  const bf16* q = Qh + ((size_t)b * S_ + i) * D_ + h * HD_;
  float qv[HD_];
#pragma unroll
  for (int d = 0; d < HD_; ++d) qv[d] = (float)q[d];
  float* outp = Qbias + ((size_t)(b * H_ + h) * S_ + i) * NB_;
  for (int r = 0; r < NB_; ++r) {
    float s = 0.f;
#pragma unroll
    for (int d = 0; d < HD_; ++d) s += qv[d] * key_bias[r * HD_ + d];
    outp[r] = s;
  }
}

// ---------------- bf16 WMMA GEMM: C[M,N] = A[M,K] @ Wt[N,K]^T ----------------
// Double-buffered LDS staging via Tensor Data Mover (TENSORcnt), wave 0 issues.
// outB != nullptr : write (acc*scale) as bf16
// outF != nullptr : write acc + resid as f32
__global__ __launch_bounds__(256) void gemm_bf16_kernel(const bf16* __restrict__ A,
                                                        const bf16* __restrict__ Wt,
                                                        float* __restrict__ outF,
                                                        bf16* __restrict__ outB,
                                                        const float* __restrict__ resid,
                                                        float scale, int K, int N) {
  __shared__ bf16 At[2][128][48];
  __shared__ bf16 Bt[2][64][48];
  const int tid = threadIdx.x;
  const int wave = tid >> 5, lane = tid & 31;
  const int ln = lane & 15, hi = lane >> 4;
  const int m0 = blockIdx.x * 128, n0 = blockIdx.y * 64;
  const int wr = wave & 3, wc = wave >> 2;
  const int koA = hi ? 16 : 0;  // A frag: lanes16-31 hold K=16..31
  const int koB = hi ? 16 : 0;  // B frag: lanes16-31 hold K=16..31

  const uint32_t aBase = (uint32_t)(uintptr_t)&At[0][0][0];
  const uint32_t bBase = (uint32_t)(uintptr_t)&Bt[0][0][0];
  const uint32_t aBufSz = 128u * 48u * 2u;
  const uint32_t bBufSz = 64u * 48u * 2u;

  floatx8 acc[2][2];
#pragma unroll
  for (int mi = 0; mi < 2; ++mi)
#pragma unroll
    for (int ni = 0; ni < 2; ++ni) acc[mi][ni] = (floatx8){0,0,0,0,0,0,0,0};

  // one K-slice = two TDM descriptors (A 128x32, B 64x32), issued by wave 0
  auto issue_tile = [&](int buf, int kk) {
    tdm_load_2d_bf16(aBase + (uint32_t)buf * aBufSz, A + (size_t)m0 * K + kk,
                     32u, 128u, (uint32_t)K);
    tdm_load_2d_bf16(bBase + (uint32_t)buf * bBufSz, Wt + (size_t)n0 * K + kk,
                     32u, 64u, (uint32_t)K);
  };

  const int NIT = K / 32;
  if (wave == 0) issue_tile(0, 0);
  for (int it = 0; it < NIT; ++it) {
    const int cur = it & 1;
    if (wave == 0) {
      if (it + 1 < NIT) {
        issue_tile(cur ^ 1, (it + 1) * 32);
        __builtin_amdgcn_s_wait_tensorcnt(2);  // current slice landed in LDS
      } else {
        __builtin_amdgcn_s_wait_tensorcnt(0);
      }
    }
    __syncthreads();

    const bf16 (*Ac)[48] = At[cur];
    const bf16 (*Bc)[48] = Bt[cur];
    bf16x16 aF[2], bF[2];
#pragma unroll
    for (int mi = 0; mi < 2; ++mi) {
      const int m = wr * 32 + mi * 16 + ln;
      aF[mi] = cat8(*(const bf16x8*)&Ac[m][koA], *(const bf16x8*)&Ac[m][koA + 8]);
    }
#pragma unroll
    for (int ni = 0; ni < 2; ++ni) {
      const int n = wc * 32 + ni * 16 + ln;
      bF[ni] = cat8(*(const bf16x8*)&Bc[n][koB], *(const bf16x8*)&Bc[n][koB + 8]);
    }
#pragma unroll
    for (int mi = 0; mi < 2; ++mi)
#pragma unroll
      for (int ni = 0; ni < 2; ++ni) acc[mi][ni] = wmma_bf16(aF[mi], bF[ni], acc[mi][ni]);
    __syncthreads();
  }

  if (outF) {
#pragma unroll
    for (int mi = 0; mi < 2; ++mi)
#pragma unroll
      for (int ni = 0; ni < 2; ++ni)
#pragma unroll
        for (int v = 0; v < 8; ++v) {
          const int gm = m0 + wr * 32 + mi * 16 + v + 8 * hi;
          const int gn = n0 + wc * 32 + ni * 16 + ln;
          const size_t o = (size_t)gm * N + gn;
          outF[o] = acc[mi][ni][v] + resid[o];
        }
  } else {
#pragma unroll
    for (int mi = 0; mi < 2; ++mi)
#pragma unroll
      for (int ni = 0; ni < 2; ++ni)
#pragma unroll
        for (int v = 0; v < 8; ++v) {
          const int gm = m0 + wr * 32 + mi * 16 + v + 8 * hi;
          const int gn = n0 + wc * 32 + ni * 16 + ln;
          outB[(size_t)gm * N + gn] = (bf16)(acc[mi][ni][v] * scale);
        }
  }
}

// ---------------- fused attention: logits -> online softmax -> probs + ctx ----------------
__global__ __launch_bounds__(128) void attn_kernel(const bf16* __restrict__ Qh,
                                                   const bf16* __restrict__ Kh,
                                                   const bf16* __restrict__ Vt,
                                                   const float* __restrict__ Qbias,
                                                   const float* __restrict__ vbias,
                                                   float* __restrict__ probs,
                                                   bf16* __restrict__ ctx) {
  __shared__ float QbL[64 * NB_];
  __shared__ float vbL[NB_ * HD_];
  __shared__ float pbL[64 * NB_];
  __shared__ bf16  pL[4][16 * 32];

  const int tid = threadIdx.x;
  const int wave = tid >> 5, lane = tid & 31;
  const int ln = lane & 15, hi = lane >> 4;

  const int blk = blockIdx.x;
  const int i_blk = blk & 15;          // S/64 = 16 row-blocks
  const int bh = blk >> 4;
  const int h = bh % H_, b = bh / H_;
  const int i0 = i_blk * 64;

  for (int idx = tid; idx < 64 * NB_; idx += 128) {
    const int row = idx / NB_, r = idx % NB_;
    QbL[idx] = Qbias[((size_t)(b * H_ + h) * S_ + i0 + row) * NB_ + r];
    pbL[idx] = 0.f;
  }
  for (int idx = tid; idx < NB_ * HD_; idx += 128) vbL[idx] = vbias[idx];
  __syncthreads();

  // Q fragments for this wave's 16 rows (cached for the whole kernel)
  const bf16* qrow = Qh + ((size_t)b * S_ + i0 + wave * 16 + ln) * D_ + h * HD_;
  const int koA = hi ? 16 : 0;
  const int koB = hi ? 16 : 0;
  const bf16x16 qa0 = cat8(*(const bf16x8*)(qrow + koA),      *(const bf16x8*)(qrow + koA + 8));
  const bf16x16 qa1 = cat8(*(const bf16x8*)(qrow + 32 + koA), *(const bf16x8*)(qrow + 32 + koA + 8));

  const bf16* kcol = Kh + ((size_t)b * S_ + ln) * D_ + h * HD_ + koB;  // + jt*16*D_

  auto logits_tile = [&](int jt) -> floatx8 {
    const bf16* kr = kcol + (size_t)jt * 16 * D_;
    const bf16x16 k0 = cat8(*(const bf16x8*)(kr),      *(const bf16x8*)(kr + 8));
    const bf16x16 k1 = cat8(*(const bf16x8*)(kr + 32), *(const bf16x8*)(kr + 40));
    floatx8 acc = (floatx8){0,0,0,0,0,0,0,0};
    acc = wmma_bf16(qa0, k0, acc);
    acc = wmma_bf16(qa1, k1, acc);
    const int j = jt * 16 + ln;
#pragma unroll
    for (int v = 0; v < 8; ++v) {
      const int rl = wave * 16 + v + 8 * hi;
      const int dji = j - (i0 + rl);
      const int r = min(max(dji, -CLIP_), CLIP_) + CLIP_;
      acc[v] += QbL[rl * NB_ + r];
    }
    return acc;
  };

  // ---- sweep 1: online row max & sum (logits discarded, recomputed later) ----
  float mrow[8], lrow[8];
#pragma unroll
  for (int v = 0; v < 8; ++v) { mrow[v] = -3.0e38f; lrow[v] = 0.f; }
  for (int jt = 0; jt < S_ / 16; ++jt) {
    if (jt + 2 < S_ / 16) __builtin_prefetch(kcol + (size_t)(jt + 2) * 16 * D_, 0, 1);
    const floatx8 acc = logits_tile(jt);
#pragma unroll
    for (int v = 0; v < 8; ++v) {
      const float tmax = rmax16(acc[v]);
      const float mn = fmaxf(mrow[v], tmax);
      const float p = __expf(acc[v] - mn);
      const float ts = rsum16(p);
      lrow[v] = lrow[v] * __expf(mrow[v] - mn) + ts;
      mrow[v] = mn;
    }
  }
  float invl[8];
#pragma unroll
  for (int v = 0; v < 8; ++v) invl[v] = 1.0f / lrow[v];

  // ---- sweep 2: normalized probs -> global, pb histogram, ctx += P @ V ----
  floatx8 accc[4];
#pragma unroll
  for (int nt = 0; nt < 4; ++nt) accc[nt] = (floatx8){0,0,0,0,0,0,0,0};
  bf16* pw = &pL[wave][0];
  const bf16* vcol = Vt + ((size_t)b * D_ + h * HD_ + ln) * S_ + koB;  // + nt*16*S_ + jc*32

  for (int jc = 0; jc < S_ / 32; ++jc) {
    if (jc + 1 < S_ / 32) {
      __builtin_prefetch(kcol + (size_t)(jc * 2 + 2) * 16 * D_, 0, 1);
      __builtin_prefetch(vcol + (size_t)(jc + 1) * 32, 0, 1);
    }
#pragma unroll
    for (int t = 0; t < 2; ++t) {
      const int jt = jc * 2 + t;
      const floatx8 acc = logits_tile(jt);
      float pv[8];
#pragma unroll
      for (int v = 0; v < 8; ++v) pv[v] = __expf(acc[v] - mrow[v]) * invl[v];

      const size_t pbase = (((size_t)h * B_ + b) * S_ + i0 + wave * 16) * S_ + jt * 16 + ln;
#pragma unroll
      for (int v = 0; v < 8; ++v) probs[pbase + (size_t)(v + 8 * hi) * S_] = pv[v];
#pragma unroll
      for (int v = 0; v < 8; ++v) pw[(v + 8 * hi) * 32 + t * 16 + ln] = (bf16)pv[v];

      // pb[i][rel] accumulation (rel saturates away from the diagonal)
      const int jlo = jt * 16;
      const int ilo = i0 + wave * 16;
      if (jlo - (ilo + 15) >= CLIP_) {
#pragma unroll
        for (int v = 0; v < 8; ++v) {
          const float s = rsum16(pv[v]);
          if (ln == 0) atomicAdd(&pbL[(wave * 16 + v + 8 * hi) * NB_ + 2 * CLIP_], s);
        }
      } else if (ilo - (jlo + 15) >= CLIP_) {
#pragma unroll
        for (int v = 0; v < 8; ++v) {
          const float s = rsum16(pv[v]);
          if (ln == 0) atomicAdd(&pbL[(wave * 16 + v + 8 * hi) * NB_ + 0], s);
        }
      } else {
        const int j = jlo + ln;
#pragma unroll
        for (int v = 0; v < 8; ++v) {
          const int rl = wave * 16 + v + 8 * hi;
          const int r = min(max(j - (i0 + rl), -CLIP_), CLIP_) + CLIP_;
          atomicAdd(&pbL[rl * NB_ + r], pv[v]);
        }
      }
    }
    asm volatile("s_wait_dscnt 0x0" ::: "memory");  // per-wave LDS RAW fence
    // repack P (16x32) into WMMA A-fragment layout
    const bf16x16 pa = cat8(*(const bf16x8*)(pw + ln * 32 + koA),
                            *(const bf16x8*)(pw + ln * 32 + koA + 8));
#pragma unroll
    for (int nt = 0; nt < 4; ++nt) {
      const bf16* vr = vcol + (size_t)nt * 16 * S_ + jc * 32;
      const bf16x16 bv = cat8(*(const bf16x8*)(vr), *(const bf16x8*)(vr + 8));
      accc[nt] = wmma_bf16(pa, bv, accc[nt]);
    }
  }

  asm volatile("s_wait_dscnt 0x0" ::: "memory");
  // epilogue: ctx += pb @ value_bias, store bf16 [B,S,D]
#pragma unroll
  for (int nt = 0; nt < 4; ++nt)
#pragma unroll
    for (int v = 0; v < 8; ++v) {
      const int rl = wave * 16 + v + 8 * hi;
      const int d = nt * 16 + ln;
      float bias = 0.f;
#pragma unroll
      for (int r = 0; r < NB_; ++r) bias += pbL[rl * NB_ + r] * vbL[r * HD_ + d];
      ctx[((size_t)b * S_ + i0 + rl) * D_ + h * HD_ + d] = (bf16)(accc[nt][v] + bias);
    }
}

// ----------------------------------------------------------------------------
extern "C" void kernel_launch(void* const* d_in, const int* in_sizes, int n_in,
                              void* d_out, int out_size, void* d_ws, size_t ws_size,
                              hipStream_t stream) {
  (void)in_sizes; (void)n_in; (void)out_size; (void)ws_size;
  const float* x  = (const float*)d_in[0];
  const float* Wq = (const float*)d_in[1];
  const float* Wk = (const float*)d_in[2];
  const float* Wv = (const float*)d_in[3];
  const float* kb = (const float*)d_in[4];
  const float* vb = (const float*)d_in[5];
  const float* Wp = (const float*)d_in[6];

  float* answer = (float*)d_out;                       // [B,S,D]
  float* probs  = answer + (size_t)B_ * S_ * D_;       // [H,B,S,S]

  char* ws = (char*)d_ws;
  auto alloc = [&](size_t bytes) { char* p = ws; ws += (bytes + 255) & ~(size_t)255; return p; };
  bf16* xb    = (bf16*)alloc((size_t)B_ * S_ * D_ * 2);
  bf16* wqt   = (bf16*)alloc((size_t)D_ * D_ * 2);
  bf16* wkt   = (bf16*)alloc((size_t)D_ * D_ * 2);
  bf16* wvt   = (bf16*)alloc((size_t)D_ * D_ * 2);
  bf16* wpt   = (bf16*)alloc((size_t)D_ * D_ * 2);
  bf16* qh    = (bf16*)alloc((size_t)B_ * S_ * D_ * 2);
  bf16* kh    = (bf16*)alloc((size_t)B_ * S_ * D_ * 2);
  bf16* vh    = (bf16*)alloc((size_t)B_ * S_ * D_ * 2);
  bf16* vt    = (bf16*)alloc((size_t)B_ * S_ * D_ * 2);
  float* qbia = (float*)alloc((size_t)B_ * H_ * S_ * NB_ * 4);
  bf16* ctxb  = (bf16*)alloc((size_t)B_ * S_ * D_ * 2);

  const int nX = B_ * S_ * D_;
  cvt_f32_bf16_kernel<<<(nX + 255) / 256, 256, 0, stream>>>(x, xb, nX);

  dim3 tb(32, 8);
  dim3 tg(D_ / 32, D_ / 32);
  wt_convert_kernel<<<tg, tb, 0, stream>>>(Wq, wqt, D_, D_);
  wt_convert_kernel<<<tg, tb, 0, stream>>>(Wk, wkt, D_, D_);
  wt_convert_kernel<<<tg, tb, 0, stream>>>(Wv, wvt, D_, D_);
  wt_convert_kernel<<<tg, tb, 0, stream>>>(Wp, wpt, D_, D_);

  dim3 gg(B_ * S_ / 128, D_ / 64);
  gemm_bf16_kernel<<<gg, 256, 0, stream>>>(xb, wqt, nullptr, qh, nullptr, 0.125f, D_, D_);
  gemm_bf16_kernel<<<gg, 256, 0, stream>>>(xb, wkt, nullptr, kh, nullptr, 1.0f, D_, D_);
  gemm_bf16_kernel<<<gg, 256, 0, stream>>>(xb, wvt, nullptr, vh, nullptr, 1.0f, D_, D_);

  vt_kernel<<<dim3(S_ / 32, D_ / 32, B_), tb, 0, stream>>>(vh, vt);
  qbias_kernel<<<(B_ * S_ * H_ + 255) / 256, 256, 0, stream>>>(qh, kb, qbia);

  attn_kernel<<<B_ * H_ * (S_ / 64), 128, 0, stream>>>(qh, kh, vt, qbia, vb, probs, ctxb);

  gemm_bf16_kernel<<<gg, 256, 0, stream>>>(ctxb, wpt, answer, nullptr, x, 1.0f, D_, D_);
}
